// GNNModel_55946243998129
// MI455X (gfx1250) — matrix operand order repaired and checked by reading, not compiled, not run on hardware
//
#include <hip/hip_runtime.h>
#include <hip/hip_bf16.h>

// ---------------- types for WMMA fragments ----------------
typedef __bf16         v16bf  __attribute__((ext_vector_type(16)));
typedef float          v8f    __attribute__((ext_vector_type(8)));
typedef unsigned int   uint4v __attribute__((ext_vector_type(4)));
typedef unsigned short us4    __attribute__((ext_vector_type(4)));
typedef unsigned short us8    __attribute__((ext_vector_type(8)));

union FragU { uint4v u[2]; v16bf h; };

__device__ __forceinline__ unsigned short f2bf(float f) {
  unsigned int u = __float_as_uint(f);
  unsigned int r = u + 0x7FFFu + ((u >> 16) & 1u);   // round-to-nearest-even
  return (unsigned short)(r >> 16);
}

// ---------------- utility kernels ----------------
__global__ void zero_f32(float* __restrict__ p, long long n) {
  long long i = (long long)blockIdx.x * blockDim.x + threadIdx.x;
  if (i < n) p[i] = 0.0f;
}

__global__ void degrees_kernel(const int* __restrict__ src, const int* __restrict__ dst,
                               float* __restrict__ degOut, float* __restrict__ degIn, int E) {
  int e = blockIdx.x * blockDim.x + threadIdx.x;
  if (e < E) {
    unsafeAtomicAdd(&degOut[src[e]], 1.0f);   // global_atomic_add_f32
    unsafeAtomicAdd(&degIn[dst[e]], 1.0f);
  }
}

// Xh[node][f] = bf16( X[node][f] * rsqrt(max(degOut[node],1)) ), 4 feats/thread
__global__ void scale_cvt(const float* __restrict__ X, const float* __restrict__ deg,
                          unsigned short* __restrict__ Xh, int cshift) {
  size_t t = (size_t)blockIdx.x * blockDim.x + threadIdx.x;
  size_t node = t >> cshift;
  int c = (int)(t & ((1u << cshift) - 1u));
  int F = 4 << cshift;
  float nrm = rsqrtf(fmaxf(deg[node], 1.0f));
  float4 v = *((const float4*)(X + node * (size_t)F) + c);
  us4 o;
  o[0] = f2bf(v.x * nrm); o[1] = f2bf(v.y * nrm);
  o[2] = f2bf(v.z * nrm); o[3] = f2bf(v.w * nrm);
  *(us4*)(Xh + node * (size_t)F + 4 * (size_t)c) = o;
}

// Wt[n][k] = bf16(W[k][n])  (W is K x N row-major); launch <<<K, N>>>
__global__ void transpose_cvt(const float* __restrict__ W, unsigned short* __restrict__ Wt,
                              int K, int N) {
  int k = blockIdx.x, n = threadIdx.x;
  Wt[(size_t)n * K + k] = f2bf(W[(size_t)k * N + n]);
}

// ---------------- WMMA GEMM: C[M x N] = A[M x K] * Bt[N x K]^T ----------------
// block = 256 threads = 8 waves; block owns a 16-row strip and ALL N columns.
// A strip (16 x K bf16, contiguous in memory) is staged once into LDS; each
// wave computes NT 16x16 tiles (cols wave*16 + j*128) reusing one A fragment.
// M must be a multiple of 16, N == NT*128, K a multiple of 32.
template <int NT>
__global__ void gemm_wmma_bf16(const unsigned short* __restrict__ A,
                               const unsigned short* __restrict__ Bt,
                               float* __restrict__ C, int K, int N) {
  __shared__ __align__(16) unsigned short As[16 * 512];   // up to K=512 (16 KB)

  const int tid  = threadIdx.x;
  const int lane = tid & 31;
  const int wave = tid >> 5;
  const int l15  = lane & 15;
  const int hi   = lane >> 4;                 // 0 or 1
  const int row0 = blockIdx.x * 16;

  // ---- cooperative stage of the contiguous 16 x K strip into LDS ----
  const us8* gsrc = (const us8*)(A + (size_t)row0 * K);
  us8* ldst = (us8*)As;
  const int chunks = (16 * K) >> 3;           // multiple of 256
  for (int c2 = tid; c2 < chunks; c2 += 256) ldst[c2] = gsrc[c2];
  __syncthreads();

  // A frag (from LDS): elems 0..7 -> K = kb + 8*hi, elems 8..15 -> kb + 16 + 8*hi
  const unsigned short* aRow = As + l15 * K + 8 * hi;
  // B frag: elems 0..7 -> K = kb + 8*hi, elems 8..15 -> kb + 16 + 8*hi for column n
  const unsigned short* bRow0 = Bt + (size_t)(wave * 16 + l15) * K + 8 * hi;
  const unsigned short* bRow1 = bRow0 + (size_t)128 * K;   // used when NT == 2

  v8f acc0 = {};
  v8f acc1 = {};
  for (int kb = 0; kb < K; kb += 32) {
    FragU a, b0, b1;
    a.u[0]  = *(const uint4v*)(aRow + kb);          // ds_load_b128
    a.u[1]  = *(const uint4v*)(aRow + kb + 16);
    b0.u[0] = *(const uint4v*)(bRow0 + kb);         // global_load_b128
    b0.u[1] = *(const uint4v*)(bRow0 + kb + 16);
    acc0 = __builtin_amdgcn_wmma_f32_16x16x32_bf16(false, a.h, false, b0.h,
                                                   (short)0, acc0, false, false);
    if (NT == 2) {
      b1.u[0] = *(const uint4v*)(bRow1 + kb);
      b1.u[1] = *(const uint4v*)(bRow1 + kb + 16);
      acc1 = __builtin_amdgcn_wmma_f32_16x16x32_bf16(false, a.h, false, b1.h,
                                                     (short)0, acc1, false, false);
    }
  }

  // D: VGPR r -> row = row0 + r + 8*hi, col = col0 + l15
  float* c0 = C + (size_t)(row0 + 8 * hi) * N + wave * 16 + l15;
  #pragma unroll
  for (int r = 0; r < 8; ++r) c0[(size_t)r * N] = acc0[r];
  if (NT == 2) {
    float* c1 = c0 + 128;
    #pragma unroll
    for (int r = 0; r < 8; ++r) c1[(size_t)r * N] = acc1[r];
  }
}

// ---------------- edge-parallel scatter: agg[dst] += H[src], 4 feats/thread ----------------
__global__ void scatter_add(const float* __restrict__ H, const int* __restrict__ src,
                            const int* __restrict__ dst, float* __restrict__ agg, int cshift) {
  size_t t = (size_t)blockIdx.x * blockDim.x + threadIdx.x;
  size_t e = t >> cshift;
  int c = (int)(t & ((1u << cshift) - 1u));
  int F = 4 << cshift;
  int s = src[e], d = dst[e];
  float4 v = *((const float4*)(H + (size_t)s * F) + c);
  float* ap = agg + (size_t)d * F + 4 * (size_t)c;
  unsafeAtomicAdd(ap + 0, v.x);
  unsafeAtomicAdd(ap + 1, v.y);
  unsafeAtomicAdd(ap + 2, v.z);
  unsafeAtomicAdd(ap + 3, v.w);
}

// Xh2[n][f] = bf16( relu(agg[n][f]*normIn[n] + b[f]) * normOut[n] )
__global__ void fuse_relu_scale(const float* __restrict__ agg, const float* __restrict__ degIn,
                                const float* __restrict__ degOut, const float* __restrict__ b,
                                unsigned short* __restrict__ Xh, int cshift) {
  size_t t = (size_t)blockIdx.x * blockDim.x + threadIdx.x;
  size_t node = t >> cshift;
  int c = (int)(t & ((1u << cshift) - 1u));
  int F = 4 << cshift;
  float ni = rsqrtf(fmaxf(degIn[node], 1.0f));
  float no = rsqrtf(fmaxf(degOut[node], 1.0f));
  float4 v  = *((const float4*)(agg + node * (size_t)F) + c);
  float4 bb = *((const float4*)b + c);
  us4 o;
  o[0] = f2bf(fmaxf(v.x * ni + bb.x, 0.0f) * no);
  o[1] = f2bf(fmaxf(v.y * ni + bb.y, 0.0f) * no);
  o[2] = f2bf(fmaxf(v.z * ni + bb.z, 0.0f) * no);
  o[3] = f2bf(fmaxf(v.w * ni + bb.w, 0.0f) * no);
  *(us4*)(Xh + node * (size_t)F + 4 * (size_t)c) = o;
}

// out[n][f] = agg[n][f]*normIn[n] + b[f]
__global__ void final_bias(const float* __restrict__ agg, const float* __restrict__ degIn,
                           const float* __restrict__ b, float* __restrict__ out, int cshift) {
  size_t t = (size_t)blockIdx.x * blockDim.x + threadIdx.x;
  size_t node = t >> cshift;
  int c = (int)(t & ((1u << cshift) - 1u));
  int F = 4 << cshift;
  float ni = rsqrtf(fmaxf(degIn[node], 1.0f));
  float4 v  = *((const float4*)(agg + node * (size_t)F) + c);
  float4 bb = *((const float4*)b + c);
  float4 o;
  o.x = v.x * ni + bb.x; o.y = v.y * ni + bb.y;
  o.z = v.z * ni + bb.z; o.w = v.w * ni + bb.w;
  *((float4*)(out + node * (size_t)F) + c) = o;
}

// ---------------- host-side launch ----------------
extern "C" void kernel_launch(void* const* d_in, const int* in_sizes, int n_in,
                              void* d_out, int out_size, void* d_ws, size_t ws_size,
                              hipStream_t stream) {
  const int NND = 50000, E = 800000, DIN = 512, DHID = 256, DOUT = 128;

  const float* features = (const float*)d_in[0];
  const int*   src      = (const int*)d_in[1];
  const int*   dst      = (const int*)d_in[2];
  const float* W1       = (const float*)d_in[3];
  const float* b1       = (const float*)d_in[4];
  const float* W2       = (const float*)d_in[5];
  const float* b2       = (const float*)d_in[6];
  float*       out      = (float*)d_out;

  // workspace carve-up (256B aligned); buffers reused across layers
  auto al = [](size_t x) { return (x + 255) & ~(size_t)255; };
  char* base = (char*)d_ws;
  size_t off = 0;
  float* degOut = (float*)(base + off); off += al((size_t)NND * 4);
  float* degIn  = (float*)(base + off); off += al((size_t)NND * 4);
  unsigned short* Xh = (unsigned short*)(base + off); off += al((size_t)NND * DIN * 2);
  unsigned short* Wt = (unsigned short*)(base + off); off += al((size_t)DIN * DHID * 2);
  float* H   = (float*)(base + off); off += al((size_t)NND * DHID * 4);
  float* AGG = (float*)(base + off); off += al((size_t)NND * DHID * 4);

  dim3 blk(256);

  // degrees
  zero_f32<<<(NND + 255) / 256, blk, 0, stream>>>(degOut, NND);
  zero_f32<<<(NND + 255) / 256, blk, 0, stream>>>(degIn, NND);
  degrees_kernel<<<(E + 255) / 256, blk, 0, stream>>>(src, dst, degOut, degIn, E);

  // ---- layer 1 ----
  zero_f32<<<(size_t)NND * DHID / 256, blk, 0, stream>>>(AGG, (long long)NND * DHID);
  scale_cvt<<<(size_t)NND * (DIN / 4) / 256, blk, 0, stream>>>(features, degOut, Xh, 7);
  transpose_cvt<<<DIN, DHID, 0, stream>>>(W1, Wt, DIN, DHID);
  gemm_wmma_bf16<2><<<NND / 16, blk, 0, stream>>>(Xh, Wt, H, DIN, DHID);
  scatter_add<<<(size_t)E * (DHID / 4) / 256, blk, 0, stream>>>(H, src, dst, AGG, 6);
  fuse_relu_scale<<<(size_t)NND * (DHID / 4) / 256, blk, 0, stream>>>(AGG, degIn, degOut, b1, Xh, 6);

  // ---- layer 2 (reuse Wt, H, AGG) ----
  transpose_cvt<<<DHID, DOUT, 0, stream>>>(W2, Wt, DHID, DOUT);
  gemm_wmma_bf16<1><<<NND / 16, blk, 0, stream>>>(Xh, Wt, H, DHID, DOUT);
  zero_f32<<<(size_t)NND * DOUT / 256, blk, 0, stream>>>(AGG, (long long)NND * DOUT);
  scatter_add<<<(size_t)E * (DOUT / 4) / 256, blk, 0, stream>>>(H, src, dst, AGG, 5);
  final_bias<<<(size_t)NND * (DOUT / 4) / 256, blk, 0, stream>>>(AGG, degIn, b2, out, 5);
}
